// USEMemory_12317966205131
// MI455X (gfx1250) — matrix operand order repaired and checked by reading, not compiled, not run on hardware
//
#include <hip/hip_runtime.h>
#include <hip/hip_bf16.h>

// ---------------------------------------------------------------------------
// Shapes: B=8, N=8, C=2048, H=W=14 -> HW=196; R=1764 rows/batch; NKV=1536.
// Pipeline (all bulk FLOPs through v_wmma_f32_16x16x32_bf16):
//   1. pack_x      : q/m (NCHW fp32) -> X bf16 (B*1764, 2048)
//   2. pack_w      : [Wk|Wv] -> WT bf16 (1536,2048) transposed + bias
//   3. gemm_nt     : KV bf16 = X @ WT^T + bias   (89 GFLOP: the bulk)
//   4. transpose_vm: VmT bf16 (B,1024,1568)
//   5. gemm_nt     : S fp32 = Kq @ Km^T per batch
//   6. softmax     : column softmax (axis=q) -> attn bf16
//   7. gemm_nt     : Vam fp32 = attn @ VmT^T per batch
//   8..10          : means, MLP gate, gated NCHW output
// GEMM: triple-buffered global_load_async_to_lds_b128 (2 chunks in flight,
// branch-free steady-state s_wait_asynccnt 0x4 with peeled final chunk),
// one barrier per K-chunk, 8 WMMAs per wave per chunk (wave 32x64, block
// 128x128), rotation-based buffer indices (no modulo in the hot loop).
// ---------------------------------------------------------------------------

typedef __attribute__((ext_vector_type(16))) __bf16        v16bf;
typedef __attribute__((ext_vector_type(8)))  float         v8f;
typedef __attribute__((ext_vector_type(4))) unsigned int   u32x4;

#define BB   8
#define NN   8
#define CC   2048
#define HW   196
#define RR   1764
#define PP   1568
#define DKk  512
#define DVv  1024
#define NKV  1536

static __device__ __forceinline__ v16bf make_frag(u32x4 lo, u32x4 hi) {
  union { u32x4 u[2]; v16bf v; } x;
  x.u[0] = lo; x.u[1] = hi;
  return x.v;
}

// --------------------------- 1. pack X (transpose + cvt) -------------------
__global__ void pack_x(const float* __restrict__ q, const float* __restrict__ m,
                       __bf16* __restrict__ X) {
  __shared__ float tile[32][33];
  const int b  = blockIdx.z;
  const int c0 = blockIdx.x * 32;
  const int r0 = blockIdx.y * 32;
  for (int yy = threadIdx.y; yy < 32; yy += 8) {
    const int c = c0 + yy;
    const int r = r0 + threadIdx.x;
    float v = 0.0f;
    if (r < RR) {
      if (r < HW) {
        v = q[((size_t)b * CC + c) * HW + r];
      } else {
        const int rr = r - HW;
        const int nn = rr / HW, s = rr % HW;
        v = m[(((size_t)b * NN + nn) * CC + c) * HW + s];
      }
    }
    tile[yy][threadIdx.x] = v;
  }
  __syncthreads();
  for (int yy = threadIdx.y; yy < 32; yy += 8) {
    const int r = r0 + yy;
    const int c = c0 + threadIdx.x;
    if (r < RR) X[((size_t)b * RR + r) * CC + c] = (__bf16)tile[threadIdx.x][yy];
  }
}

// --------------------------- 2. pack W (transpose + cvt) -------------------
__global__ void pack_w(const float* __restrict__ Wk, const float* __restrict__ bk,
                       const float* __restrict__ Wv, const float* __restrict__ bv,
                       __bf16* __restrict__ WT, float* __restrict__ bias) {
  const int n = blockIdx.x;
  for (int k = threadIdx.x; k < CC; k += 256) {
    float v = (n < DKk) ? Wk[(size_t)k * DKk + n]
                        : Wv[(size_t)k * DVv + (n - DKk)];
    WT[(size_t)n * CC + k] = (__bf16)v;
  }
  if (threadIdx.x == 0) bias[n] = (n < DKk) ? bk[n] : bv[n - DKk];
}

// --------------------------- 3/5/7. bf16 WMMA GEMM (NT) --------------------
// D(MxN) = A(MxK) * Bt(NxK)^T [+bias]; A/Bt bf16 row-major, K-contiguous.
// Block tile 128x128, 8 waves, wave tile 32x64 -> 8 WMMAs/chunk/wave. KC=32.
// Async DMA staging, triple-buffered; edge rows/cols address-clamped (their
// outputs are store-masked, so duplicate loads are benign).
__global__ void gemm_nt(const __bf16* __restrict__ A, const __bf16* __restrict__ Bt,
                        const float* __restrict__ bias,
                        float* __restrict__ Df32, __bf16* __restrict__ Dbf16,
                        int M, int Nn, int K, int lda, int ldb, int ldd,
                        long long strideA, long long strideB, long long strideD) {
  // row stride = 5 * uint4 (80 B): keeps 16B alignment, kills bank conflicts
  __shared__ u32x4 As[3][128 * 5];   // 30 KB
  __shared__ u32x4 Bs[3][128 * 5];   // 30 KB

  const int t  = threadIdx.x;
  const int bz = blockIdx.z;
  A  += (size_t)bz * strideA;
  Bt += (size_t)bz * strideB;
  const size_t dOff = (size_t)bz * strideD;

  const int mBase = blockIdx.x * 128;
  const int nBase = blockIdx.y * 128;

  // staging roles: 128 rows x 4 uint4 each for A and B; 2+2 asyncs per thread
  const int srow = t >> 1;              // 0..127
  const int sk0  = (t & 1) * 2;         // uint4 idx 0 or 2
  const int gmA  = min(mBase + srow, M - 1);   // clamp: no divergent guards
  const int gnB  = min(nBase + srow, Nn - 1);
  const __bf16* agp = A  + (size_t)gmA * lda + sk0 * 8;
  const __bf16* bgp = Bt + (size_t)gnB * ldb + sk0 * 8;

  // wave/fragment roles
  const int lane = t & 31;
  const int wv   = t >> 5;
  const int msub = (wv & 3) << 5;       // 0,32,64,96
  const int nsub = (wv >> 2) << 6;      // 0,64
  const int l16  = lane & 15;
  const int sel  = lane >> 4;

  v8f acc[8] = {};

  // Async DMA: per ISA 10.2 the low 32 bits of a generic LDS pointer are the
  // raw LDS byte offset. 4 ops/chunk.
#define ISSUE_CHUNK(bufi, kc)                                                  \
  do {                                                                         \
    unsigned la = (unsigned)(uintptr_t)&As[(bufi)][srow * 5 + sk0];            \
    unsigned long long ga = (unsigned long long)(uintptr_t)(agp + (kc));       \
    asm volatile("global_load_async_to_lds_b128 %0, %1, off"                   \
                 :: "v"(la), "v"(ga) : "memory");                              \
    asm volatile("global_load_async_to_lds_b128 %0, %1, off"                   \
                 :: "v"(la + 16u), "v"(ga + 16ull) : "memory");                \
    unsigned lb = (unsigned)(uintptr_t)&Bs[(bufi)][srow * 5 + sk0];            \
    unsigned long long gb = (unsigned long long)(uintptr_t)(bgp + (kc));       \
    asm volatile("global_load_async_to_lds_b128 %0, %1, off"                   \
                 :: "v"(lb), "v"(gb) : "memory");                              \
    asm volatile("global_load_async_to_lds_b128 %0, %1, off"                   \
                 :: "v"(lb + 16u), "v"(gb + 16ull) : "memory");                \
  } while (0)

  // Batch ALL fragment ds_loads first so WMMAs issue back-to-back.
  // A frag: lanes 0-15: K0-7 (idx0)+K16-23 (idx2); lanes 16-31: idx1/idx3.
#define COMPUTE_CHUNK()                                                        \
  do {                                                                         \
    const int ar0 = (msub + l16) * 5;                                          \
    const int ar1 = (msub + 16 + l16) * 5;                                     \
    const u32x4 a00 = As[cur][ar0 + sel];                                      \
    const u32x4 a01 = As[cur][ar0 + 2 + sel];                                  \
    const u32x4 a10 = As[cur][ar1 + sel];                                      \
    const u32x4 a11 = As[cur][ar1 + 2 + sel];                                  \
    u32x4 bb[8];                                                               \
    _Pragma("unroll")                                                          \
    for (int j = 0; j < 4; ++j) {                                              \
      const int bc = (nsub + j * 16 + l16) * 5 + sel;                          \
      bb[2 * j]     = Bs[cur][bc];                                             \
      bb[2 * j + 1] = Bs[cur][bc + 2];                                         \
    }                                                                          \
    const v16bf af0 = make_frag(a00, a01);                                     \
    const v16bf af1 = make_frag(a10, a11);                                     \
    _Pragma("unroll")                                                          \
    for (int j = 0; j < 4; ++j) {                                              \
      const v16bf bfj = make_frag(bb[2 * j], bb[2 * j + 1]);                   \
      acc[j]     = __builtin_amdgcn_wmma_f32_16x16x32_bf16(                    \
          false, af0, false, bfj, (short)0, acc[j], false, false);             \
      acc[4 + j] = __builtin_amdgcn_wmma_f32_16x16x32_bf16(                    \
          false, af1, false, bfj, (short)0, acc[4 + j], false, false);         \
    }                                                                          \
  } while (0)

  ISSUE_CHUNK(0, 0);
  if (32 < K) ISSUE_CHUNK(1, 32);
  int cur = 0, nxt = 1, spare = 2;
  int kc = 0;
  // Steady state (branch-free wait): chunks kc and kc+32 are in flight
  // (8 ops); async completes in order, so <=4 outstanding means chunk kc
  // has fully landed.
  for (; kc + 32 < K; kc += 32) {
    asm volatile("s_wait_asynccnt 0x4" ::: "memory");
    __syncthreads();
    if (kc + 64 < K) ISSUE_CHUNK(spare, kc + 64);
    COMPUTE_CHUNK();
    const int tmp = cur; cur = nxt; nxt = spare; spare = tmp;  // rotate bufs
    // No trailing barrier: each wave's dscnt wait precedes its WMMAs, which
    // precede the next iteration's barrier; a buffer is only re-issued three
    // chunks after its last read.
  }
  // Peeled final chunk: drain everything.
  asm volatile("s_wait_asynccnt 0x0" ::: "memory");
  __syncthreads();
  COMPUTE_CHUNK();
#undef COMPUTE_CHUNK
#undef ISSUE_CHUNK

  // Epilogue: VGPR i, lane l -> M = i + 8*(l>=16), N = l&15
  #pragma unroll
  for (int h = 0; h < 2; ++h) {
    #pragma unroll
    for (int j = 0; j < 4; ++j) {
      const int n = nBase + nsub + j * 16 + l16;
      if (n < Nn) {
        const float bs = bias ? bias[n] : 0.0f;
        #pragma unroll
        for (int i = 0; i < 8; ++i) {
          const int m = mBase + msub + h * 16 + sel * 8 + i;
          if (m < M) {
            const float v = acc[h * 4 + j][i] + bs;
            if (Df32)  Df32 [dOff + (size_t)m * ldd + n] = v;
            if (Dbf16) Dbf16[dOff + (size_t)m * ldd + n] = (__bf16)v;
          }
        }
      }
    }
  }
}

// --------------------------- 4. transpose Vm -------------------------------
__global__ void transpose_vm(const __bf16* __restrict__ KV, __bf16* __restrict__ VmT) {
  __shared__ __bf16 tile[32][33];
  const int b  = blockIdx.z;
  const int p0 = blockIdx.x * 32;
  const int n0 = blockIdx.y * 32;
  for (int yy = threadIdx.y; yy < 32; yy += 8) {
    const int p = p0 + yy;
    const int n = n0 + threadIdx.x;
    __bf16 v = (__bf16)0.0f;
    if (p < PP) v = KV[((size_t)b * RR + HW + p) * NKV + DKk + n];
    tile[yy][threadIdx.x] = v;
  }
  __syncthreads();
  for (int yy = threadIdx.y; yy < 32; yy += 8) {
    const int n = n0 + yy;
    const int p = p0 + threadIdx.x;
    if (p < PP) VmT[((size_t)b * DVv + n) * PP + p] = tile[threadIdx.x][yy];
  }
}

// --------------------------- 6. softmax over axis=q ------------------------
__global__ void softmax_cols(const float* __restrict__ S, __bf16* __restrict__ attn) {
  const int b = blockIdx.y;
  const int p = blockIdx.x * 256 + threadIdx.x;
  if (p >= PP) return;
  const float* col = S + (size_t)b * HW * PP + p;
  float mx = -3.4e38f;
  for (int r = 0; r < HW; ++r) mx = fmaxf(mx, col[(size_t)r * PP]);
  float sum = 0.0f;
  for (int r = 0; r < HW; ++r) sum += __expf(col[(size_t)r * PP] - mx);
  const float inv = 1.0f / sum;
  __bf16* ac = attn + (size_t)b * HW * PP + p;
  for (int r = 0; r < HW; ++r)
    ac[(size_t)r * PP] = (__bf16)(__expf(col[(size_t)r * PP] - mx) * inv);
}

// --------------------------- 8. spatial means ------------------------------
__global__ void means(const float* __restrict__ Vam, const __bf16* __restrict__ KV,
                      float* __restrict__ gV) {
  const int b = blockIdx.y;
  const int n = blockIdx.x * 256 + threadIdx.x;
  float sa = 0.0f, sq = 0.0f;
  for (int s = 0; s < HW; ++s) {
    sa += Vam[((size_t)b * HW + s) * DVv + n];
    sq += (float)KV[((size_t)b * RR + s) * NKV + DKk + n];
  }
  const float scale = 1.0f / (float)HW;
  gV[(size_t)b * 2048 + n]       = sa * scale;
  gV[(size_t)b * 2048 + DVv + n] = sq * scale;
}

// --------------------------- 9. tiny MLP + sigmoid gate --------------------
__global__ void mlp_gate(const float* __restrict__ gV,
                         const float* __restrict__ W1, const float* __restrict__ b1,
                         const float* __restrict__ W2, const float* __restrict__ b2,
                         float* __restrict__ wgt) {
  __shared__ float xs[2048];
  __shared__ float h[1024];
  const int b = blockIdx.x;
  for (int i = threadIdx.x; i < 2048; i += 256) xs[i] = gV[(size_t)b * 2048 + i];
  __syncthreads();
  for (int j = threadIdx.x; j < 1024; j += 256) {
    float acc = b1[j];
    for (int k = 0; k < 2048; ++k) acc += xs[k] * W1[(size_t)k * 1024 + j];
    h[j] = acc;
  }
  __syncthreads();
  for (int o = threadIdx.x; o < 2048; o += 256) {
    float acc = b2[o];
    for (int k = 0; k < 1024; ++k) acc += h[k] * W2[(size_t)k * 2048 + o];
    wgt[(size_t)b * 2048 + o] = 1.0f / (1.0f + __expf(-acc));
  }
}

// --------------------------- 10. gated output (to NCHW) --------------------
__global__ void final_out(const float* __restrict__ Vam, const __bf16* __restrict__ KV,
                          const float* __restrict__ wgt, float* __restrict__ out) {
  const int b  = blockIdx.z;
  const int ch = blockIdx.y;
  const int s  = threadIdx.x;
  if (s >= HW) return;
  const float w = wgt[(size_t)b * 2048 + ch];
  float v;
  if (ch < DVv) v = Vam[((size_t)b * HW + s) * DVv + ch];
  else          v = (float)KV[((size_t)b * RR + s) * NKV + DKk + (ch - DVv)];
  out[((size_t)b * CC + ch) * HW + s] = v * w;
}

// ---------------------------------------------------------------------------
extern "C" void kernel_launch(void* const* d_in, const int* in_sizes, int n_in,
                              void* d_out, int out_size, void* d_ws, size_t ws_size,
                              hipStream_t stream) {
  (void)in_sizes; (void)n_in; (void)out_size; (void)ws_size;
  const float* q  = (const float*)d_in[0];
  const float* m  = (const float*)d_in[1];
  const float* Wk = (const float*)d_in[2];
  const float* bk = (const float*)d_in[3];
  const float* Wv = (const float*)d_in[4];
  const float* bv = (const float*)d_in[5];
  const float* W1 = (const float*)d_in[6];
  const float* b1 = (const float*)d_in[7];
  const float* W2 = (const float*)d_in[8];
  const float* b2 = (const float*)d_in[9];
  float* out = (float*)d_out;

  char* ws = (char*)d_ws;
  size_t off = 0;
  auto take = [&](size_t bytes) -> char* {
    char* p = ws + off;
    off += (bytes + 255) & ~(size_t)255;
    return p;
  };
  __bf16* Xb   = (__bf16*)take((size_t)BB * RR * CC * 2);
  __bf16* WT   = (__bf16*)take((size_t)NKV * CC * 2);
  float*  bias = (float*) take((size_t)NKV * 4);
  __bf16* KV   = (__bf16*)take((size_t)BB * RR * NKV * 2);
  __bf16* VmT  = (__bf16*)take((size_t)BB * DVv * PP * 2);
  float*  S    = (float*) take((size_t)BB * HW * PP * 4);
  __bf16* At   = (__bf16*)take((size_t)BB * HW * PP * 2);
  float*  Vam  = (float*) take((size_t)BB * HW * DVv * 4);
  float*  gV   = (float*) take((size_t)BB * 2048 * 4);
  float*  wgt  = (float*) take((size_t)BB * 2048 * 4);

  pack_x<<<dim3(CC / 32, (RR + 31) / 32, BB), dim3(32, 8), 0, stream>>>(q, m, Xb);
  pack_w<<<NKV, 256, 0, stream>>>(Wk, bk, Wv, bv, WT, bias);
  // KV = X @ WT^T + bias  (M=14112, N=1536, K=2048) -> bf16
  gemm_nt<<<dim3((BB * RR + 127) / 128, NKV / 128, 1), 256, 0, stream>>>(
      Xb, WT, bias, nullptr, KV,
      BB * RR, NKV, CC, CC, CC, NKV, 0LL, 0LL, 0LL);
  transpose_vm<<<dim3((PP + 31) / 32, DVv / 32, BB), dim3(32, 8), 0, stream>>>(KV, VmT);
  // S = Kq @ Km^T per batch (M=196, N=1568, K=512)
  gemm_nt<<<dim3((HW + 127) / 128, (PP + 127) / 128, BB), 256, 0, stream>>>(
      KV, KV + (size_t)HW * NKV, nullptr, S, nullptr,
      HW, PP, DKk, NKV, NKV, PP,
      (long long)RR * NKV, (long long)RR * NKV, (long long)HW * PP);
  softmax_cols<<<dim3((PP + 255) / 256, BB), 256, 0, stream>>>(S, At);
  // Vam = attn @ VmT^T per batch (M=196, N=1024, K=1568)
  gemm_nt<<<dim3((HW + 127) / 128, DVv / 128, BB), 256, 0, stream>>>(
      At, VmT, nullptr, Vam, nullptr,
      HW, DVv, PP, PP, PP, DVv,
      (long long)HW * PP, (long long)DVv * PP, (long long)HW * DVv);
  means<<<dim3(DVv / 256, BB), 256, 0, stream>>>(Vam, KV, gV);
  mlp_gate<<<BB, 256, 0, stream>>>(gV, W1, b1, W2, b2, wgt);
  final_out<<<dim3(1, CC, BB), 256, 0, stream>>>(Vam, KV, wgt, out);
}
